// PolicyNet_38457137169048
// MI455X (gfx1250) — matrix operand order repaired and checked by reading, not compiled, not run on hardware
//
#include <hip/hip_runtime.h>

// ---------------- problem constants ----------------
#define B_TOT   131072
#define HID     24
#define NOUT    24
#define LLIM    12
#define NSTEPS  23

// ---------------- launch shape ----------------
#define WAVES_PB   2
#define BLOCK_T    (WAVES_PB * 32)
#define ITEMS_PW   32                     // 2 WMMA M-tiles of 16 rows per wave
#define ITEMS_PB   (WAVES_PB * ITEMS_PW)
#define NBLOCKS    (B_TOT / ITEMS_PB)     // 2048

typedef _Float16 v16h __attribute__((ext_vector_type(16)));
typedef _Float16 v8h  __attribute__((ext_vector_type(8)));
typedef float    v8f  __attribute__((ext_vector_type(8)));

union V16U { v16h v; v8h h2[2]; _Float16 e[16]; };
union V8FU { v8f v; float f[8]; };
union F4U  { float4 v; float f[4]; };
union V8HU { v8h v; _Float16 e[8]; };

__device__ __forceinline__ float sigf(float x) { return 1.0f / (1.0f + __expf(-x)); }
__device__ __forceinline__ float tanh_fast(float x) {
  float t = __expf(-2.0f * x);
  return (1.0f - t) / (1.0f + t);
}

__global__ void __launch_bounds__(BLOCK_T)
policy_lstm_sample_kernel(const float* __restrict__ W_ih,
                          const float* __restrict__ W_hh,
                          const float* __restrict__ b_ih,
                          const float* __restrict__ b_hh,
                          const float* __restrict__ W_lin,
                          const float* __restrict__ b_lin,
                          const float* __restrict__ rand_u,
                          float* __restrict__ out)
{
  // per-block LDS:
  //  s_wihT : [25][96]: rows 0..23 = W_ih^T row x + (b_ih+b_hh); row 24 = bias only (x=0 step)
  //  s_blin : [24] linear-head bias (broadcast reads in phase 4)
  //  s_gates: per-wave gate pre-activations [32 items][96]
  //  s_p    : per-wave linear-head output  [32 items][32] (24 valid, padded)
  //  s_h    : per-wave hidden state, f16, [32 items][32] (K padded 24..31 = 0)
  __shared__ __align__(16) float    s_wihT[(HID + 1) * 96];
  __shared__ __align__(16) float    s_blin[NOUT];
  __shared__ __align__(16) float    s_gates[WAVES_PB][ITEMS_PW * 96];
  __shared__ __align__(16) float    s_p[WAVES_PB][ITEMS_PW * 32];
  __shared__ __align__(16) _Float16 s_h[WAVES_PB][ITEMS_PW * 32];

  const int tid    = threadIdx.x;
  const int wave   = tid >> 5;
  const int lane   = tid & 31;
  const int lane16 = lane & 15;
  const int hiHalf = lane >> 4;
  const int ak0    = hiHalf * 8;    // A layout: low lanes K{0-7,16-23}, high lanes K{8-15,24-31}
  const int bk0    = hiHalf * 16;   // unused; B uses the A order: low lanes K{0-7,16-23}, high lanes K{8-15,24-31}
  const int itemBase = blockIdx.x * ITEMS_PB + wave * ITEMS_PW;
  const int myItem   = itemBase + lane;

  float*    gbuf = s_gates[wave];
  float*    pbuf = s_p[wave];
  _Float16* hbuf = s_h[wave];

  // ---- one-time LDS tables ----
  // s_wihT[x*96 + j] = W_ih[j*24 + x] + b_ih[j] + b_hh[j]   (x<24)
  // s_wihT[24*96+ j] =                  b_ih[j] + b_hh[j]   (bias-only row for x=0)
  for (int idx = tid; idx < (HID + 1) * 96; idx += BLOCK_T) {
    int x = idx / 96, j = idx % 96;
    float v = b_ih[j] + b_hh[j];
    if (x < HID) v += W_ih[j * NOUT + x];
    s_wihT[idx] = v;
  }
  if (tid < NOUT) s_blin[tid] = b_lin[tid];
  // zero f16 hidden buffer (padding cols 24..31 stay zero forever)
  {
    uint4 z; z.x = z.y = z.z = z.w = 0u;
    uint4* dst = (uint4*)(hbuf + lane * 32);
    dst[0] = z; dst[1] = z; dst[2] = z; dst[3] = z;
  }
  __syncthreads();

  // ---- preload wave-resident WMMA B operands ----
  // gates = h @ W_hh^T  ->  B[k][n] = W_hh[n][k], 6 N-tiles of 16 features
  V16U bhh[6];
#pragma unroll
  for (int t = 0; t < 6; ++t) {
    const int j = 16 * t + lane16;           // gate feature column
#pragma unroll
    for (int e = 0; e < 16; ++e) {
      const int k = ak0 + e + ((e >> 3) << 3); // hidden index (padded to 32)
      bhh[t].e[e] = (k < HID) ? (_Float16)W_hh[j * HID + k] : (_Float16)0.0f;
    }
  }
  // p = h @ W_lin^T -> B[k][n] = W_lin[n][k], 2 N-tiles (cols 24..31 zero)
  V16U blin2[2];
#pragma unroll
  for (int t = 0; t < 2; ++t) {
    const int j = 16 * t + lane16;
#pragma unroll
    for (int e = 0; e < 16; ++e) {
      const int k = ak0 + e + ((e >> 3) << 3);
      blin2[t].e[e] = (j < NOUT && k < HID) ? (_Float16)W_lin[j * HID + k]
                                            : (_Float16)0.0f;
    }
  }
  // shared all-zero C operand for every WMMA (biases folded into LDS tables)
  V8FU zc;
#pragma unroll
  for (int v = 0; v < 8; ++v) zc.f[v] = 0.0f;

  // single-base store/load pointers (all tile offsets become DS immediates)
  float*           gstore = gbuf + (hiHalf * 8) * 96 + lane16;
  float*           pstore = pbuf + (hiHalf * 8) * 32 + lane16;
  const _Float16*  habase = hbuf + lane16 * 32 + ak0;

  // ---- per-item recurrent state (lane = item) ----
  float creg[HID];
#pragma unroll
  for (int k = 0; k < HID; ++k) creg[k] = 0.0f;
  int xirow = HID;      // W_ih^T table row; row 24 = bias-only (x = 0, first step)
  int stepIdx = 0;

#pragma unroll 1
  for (int counter = 0; counter < LLIM; ++counter) {
#pragma unroll 1
    for (int sub = 0; sub < 2; ++sub) {
      if (counter == 0 && sub == 0) continue;   // uniform across block

      // ===== phase 1: gates = h @ W_hh^T  (WMMA, C = 0) =====
      V16U amat[2];
#pragma unroll
      for (int m = 0; m < 2; ++m) {
        amat[m].h2[0] = *(const v8h*)(habase + m * 16 * 32);
        amat[m].h2[1] = *(const v8h*)(habase + m * 16 * 32 + 16);
      }
#pragma unroll
      for (int m = 0; m < 2; ++m) {
#pragma unroll
        for (int t = 0; t < 6; ++t) {
          V8FU dd;
          dd.v = __builtin_amdgcn_wmma_f32_16x16x32_f16(
              false, amat[m].v, false, bhh[t].v, (short)0, zc.v, false, false);
#pragma unroll
          for (int v = 0; v < 8; ++v)
            gstore[(16 * m + v) * 96 + 16 * t] = dd.f[v];   // one base + immediates
        }
      }
      __syncthreads();

      // ===== phase 2: per-item gate math (lane = item) =====
      // g += s_wihT[xirow]  (= one-hot W_ih column + gate biases, always applied)
      float hreg[HID];
      {
        const float* grow = gbuf + lane * 96;
        const float* wrow = s_wihT + xirow * 96;
#pragma unroll
        for (int kb = 0; kb < HID / 4; ++kb) {
          F4U iv, fv, gv, ov, wi, wf, wg, wo;
          iv.v = *(const float4*)(grow +  0 + kb * 4);
          fv.v = *(const float4*)(grow + 24 + kb * 4);
          gv.v = *(const float4*)(grow + 48 + kb * 4);
          ov.v = *(const float4*)(grow + 72 + kb * 4);
          wi.v = *(const float4*)(wrow +  0 + kb * 4);
          wf.v = *(const float4*)(wrow + 24 + kb * 4);
          wg.v = *(const float4*)(wrow + 48 + kb * 4);
          wo.v = *(const float4*)(wrow + 72 + kb * 4);
#pragma unroll
          for (int q = 0; q < 4; ++q) {
            const int k = kb * 4 + q;
            float ivx = iv.f[q] + wi.f[q];
            float fvx = fv.f[q] + wf.f[q];
            float gvx = gv.f[q] + wg.f[q];
            float ovx = ov.f[q] + wo.f[q];
            float cn = sigf(fvx) * creg[k] + sigf(ivx) * tanh_fast(gvx);
            creg[k] = cn;
            hreg[k] = sigf(ovx) * tanh_fast(cn);
          }
        }
      }
      // write h back in A-matrix f16 layout (row = item, 24 halfs; pad stays 0)
      {
        _Float16* dst = hbuf + lane * 32;
#pragma unroll
        for (int q8 = 0; q8 < 3; ++q8) {
          V8HU hp;
#pragma unroll
          for (int e = 0; e < 8; ++e) hp.e[e] = (_Float16)hreg[q8 * 8 + e];
          *(v8h*)(dst + q8 * 8) = hp.v;
        }
      }
      __syncthreads();

      // ===== phase 3: p_raw = h @ W_lin^T  (WMMA, C = 0) =====
#pragma unroll
      for (int m = 0; m < 2; ++m) {
        V16U a2;
        a2.h2[0] = *(const v8h*)(habase + m * 16 * 32);
        a2.h2[1] = *(const v8h*)(habase + m * 16 * 32 + 16);
#pragma unroll
        for (int t = 0; t < 2; ++t) {
          V8FU dd;
          dd.v = __builtin_amdgcn_wmma_f32_16x16x32_f16(
              false, a2.v, false, blin2[t].v, (short)0, zc.v, false, false);
#pragma unroll
          for (int v = 0; v < 8; ++v)
            pstore[(16 * m + v) * 32 + 16 * t] = dd.f[v];
        }
      }
      __syncthreads();

      // ===== phase 4: per-item softmax + sliced softmax + inverse-CDF sample =====
      float p[NOUT];
      {
        const float* prow = pbuf + lane * 32;
#pragma unroll
        for (int q = 0; q < NOUT / 4; ++q) {
          F4U t4, b4;
          t4.v = *(const float4*)(prow + q * 4);
          b4.v = *(const float4*)(s_blin + q * 4);    // broadcast, conflict-free
#pragma unroll
          for (int e = 0; e < 4; ++e) p[q * 4 + e] = t4.f[e] + b4.f[e];
        }
      }
      float mx = p[0];
#pragma unroll
      for (int j = 1; j < NOUT; ++j) mx = fmaxf(mx, p[j]);
      float ssum = 0.0f;
#pragma unroll
      for (int j = 0; j < NOUT; ++j) { p[j] = __expf(p[j] - mx); ssum += p[j]; }
      const float inv = 1.0f / ssum;
#pragma unroll
      for (int j = 0; j < NOUT; ++j) p[j] *= inv;

      int n, base, off;
      if (sub == 0) { n = counter;         base = NOUT - LLIM; off = NOUT - LLIM; } // dag/father
      else          { n = NOUT - LLIM - 1; base = 0;           off = 1;           } // action

      // branch-free sliced softmax (p[base+j] always in-bounds: base+10 <= 22)
      float mx2 = -1e30f;
#pragma unroll
      for (int j = 0; j < 11; ++j) {
        float pv = p[base + j];
        mx2 = (j < n) ? fmaxf(mx2, pv) : mx2;
      }
      float q2[11];
      float s2 = 0.0f;
#pragma unroll
      for (int j = 0; j < 11; ++j) {
        float ev = __expf(p[base + j] - mx2);
        ev = (j < n) ? ev : 0.0f;
        q2[j] = ev; s2 += ev;
      }
      const float inv2 = 1.0f / s2;
      const float u = rand_u[stepIdx * B_TOT + myItem];

      // branch-free inverse-CDF: first index where cumsum > u (else 0)
      float cs = 0.0f; int idx = -1; float prob = 0.0f;
#pragma unroll
      for (int j = 0; j < 11; ++j) {
        float qv = q2[j] * inv2;        // 0 for masked-off j
        cs += qv;
        bool hit = (idx < 0) & (cs > u) & (j < n);
        idx  = hit ? j  : idx;
        prob = hit ? qv : prob;
      }
      bool none = (idx < 0);
      idx  = none ? 0 : idx;            // argmax(all-false) == 0
      prob = none ? q2[0] * inv2 : prob;

      const int aval = idx + off;
      out[myItem * NOUT + stepIdx] = (float)aval;                 // actions[:, step]
      out[B_TOT * NOUT + myItem * NSTEPS + stepIdx] = prob;       // probs[:, step]
      xirow = aval;                                               // next one-hot input row
      ++stepIdx;
    }
  }
  out[myItem * NOUT + (NOUT - 1)] = 0.0f;   // actions column 23 never written
}

extern "C" void kernel_launch(void* const* d_in, const int* in_sizes, int n_in,
                              void* d_out, int out_size, void* d_ws, size_t ws_size,
                              hipStream_t stream) {
  (void)in_sizes; (void)n_in; (void)out_size; (void)d_ws; (void)ws_size;
  const float* W_ih   = (const float*)d_in[0];
  const float* W_hh   = (const float*)d_in[1];
  const float* b_ih   = (const float*)d_in[2];
  const float* b_hh   = (const float*)d_in[3];
  const float* W_lin  = (const float*)d_in[4];
  const float* b_lin  = (const float*)d_in[5];
  const float* rand_u = (const float*)d_in[6];
  float* out = (float*)d_out;
  policy_lstm_sample_kernel<<<NBLOCKS, BLOCK_T, 0, stream>>>(
      W_ih, W_hh, b_ih, b_hh, W_lin, b_lin, rand_u, out);
}
